// Attention_89060441850471
// MI455X (gfx1250) — matrix operand order repaired and checked by reading, not compiled, not run on hardware
//
#include <hip/hip_runtime.h>
#include <hip/hip_bf16.h>
#include <math.h>

// ---------------------------------------------------------------------------
// Problem constants (match the reference)
// ---------------------------------------------------------------------------
#define DIM      2048
#define N_HEADS  16
#define HEAD_DIM 128          // DIM / N_HEADS
#define BATCH    2
#define SEQ      2048
#define LDQKV    (3 * DIM)    // row stride of the qkv workspace (elements)

typedef __attribute__((ext_vector_type(16))) _Float16 v16h;
typedef __attribute__((ext_vector_type(8)))  _Float16 v8h;
typedef __attribute__((ext_vector_type(8)))  float    v8f;
typedef __attribute__((ext_vector_type(4)))  unsigned int v4u;
typedef __attribute__((ext_vector_type(8)))  int      v8i;
typedef __attribute__((ext_vector_type(4)))  int      v4i;

// ---------------------------------------------------------------------------
// WMMA wrapper: D = A(16x32 f16) * B(32x16 f16) + C(16x16 f32)
// ---------------------------------------------------------------------------
__device__ __forceinline__ v8f wmma_f16(v16h a, v16h b, v8f c) {
  return __builtin_amdgcn_wmma_f32_16x16x32_f16(
      /*neg_a=*/false, a, /*neg_b=*/false, b,
      /*c_mod=*/(short)0, c, /*reuse_a=*/false, /*reuse_b=*/false);
}

// ---------------------------------------------------------------------------
// Fragment loaders (wave32; A/B in this code's contiguous k order, not the ISA table order), f16 sources.
//
// A (16x32): lane<16 -> row M=lane,    K = 0..15
//            lane>=16 -> row M=lane-16, K = 16..31
// B (32x16): lane<16 -> col N=lane,    K = 0..15
//            lane>=16 -> col N=lane-16, K = 16..31
// C (f32):   vgpr r: lanes0-15 -> (M=r, N=lane), lanes16-31 -> (M=r+8, N=lane-16)
// ---------------------------------------------------------------------------
__device__ __forceinline__ v16h load_a_h(const _Float16* __restrict__ base, size_t ld) {
  const int lane = threadIdx.x & 31;
  const int row  = lane & 15;
  const int kb   = (lane < 16) ? 0 : 16;
  const _Float16* p = base + (size_t)row * ld + kb;
  v8h lo = *(const v8h*)p;
  v8h hi = *(const v8h*)(p + 8);
  v16h a;
#pragma unroll
  for (int i = 0; i < 8; ++i) { a[i] = lo[i]; a[8 + i] = hi[i]; }
  return a;
}

// B-frag where source is row-major N x K (i.e. B^T): one 32B contiguous load.
__device__ __forceinline__ v16h load_bT_h(const _Float16* __restrict__ base, size_t ld,
                                          int row0, int col) {
  const int lane = threadIdx.x & 31;
  const int n    = lane & 15;
  const int kb   = (lane < 16) ? 0 : 16;
  const _Float16* p = base + (size_t)(row0 + n) * ld + col + kb;
  v8h lo = *(const v8h*)p;
  v8h hi = *(const v8h*)(p + 8);
  v16h b;
#pragma unroll
  for (int i = 0; i < 8; ++i) { b[i] = lo[i]; b[8 + i] = hi[i]; }
  return b;
}

// B-frag where source is row-major K x N (V tile in LDS): strided along K.
__device__ __forceinline__ v16h load_b_kn_lds(const _Float16* __restrict__ tile, int ld, int col) {
  const int lane = threadIdx.x & 31;
  const int n    = lane & 15;
  const int kb   = (lane < 16) ? 0 : 16;
  v16h b;
#pragma unroll
  for (int i = 0; i < 16; ++i) b[i] = tile[(kb + i) * ld + col + n];
  return b;
}

// 16-lane-group reductions (wave32: xor masks 1,2,4,8 stay within each half)
__device__ __forceinline__ float red_max16(float x) {
  x = fmaxf(x, __shfl_xor(x, 1, 32));
  x = fmaxf(x, __shfl_xor(x, 2, 32));
  x = fmaxf(x, __shfl_xor(x, 4, 32));
  x = fmaxf(x, __shfl_xor(x, 8, 32));
  return x;
}
__device__ __forceinline__ float red_sum16(float x) {
  x += __shfl_xor(x, 1, 32);
  x += __shfl_xor(x, 2, 32);
  x += __shfl_xor(x, 4, 32);
  x += __shfl_xor(x, 8, 32);
  return x;
}

// ---------------------------------------------------------------------------
// Tensor Data Mover: 2D tile (tile_d0 x rows f16 elements) global -> LDS.
// D# packing per CDNA5 ISA 8.3/8.4 (group0 + group1; groups 2/3 zero => 2D).
// This toolchain's builtin is the 6-arg clang-23 form:
//   (uint32x4 g0, int32x8 g1, int32x4 g2, int32x4 g3, int32x8 extra, i32 cpol)
// ---------------------------------------------------------------------------
__device__ __forceinline__ void tdm_load_tile_f16(unsigned lds_byte_off,
                                                  const _Float16* gsrc,
                                                  unsigned tile_d0,      // elems per row
                                                  unsigned rows,         // tile_dim1
                                                  unsigned rows_avail,   // tensor_dim1 (OOB->0)
                                                  unsigned long long row_stride) { // elems
  unsigned long long ga = (unsigned long long)(uintptr_t)gsrc;
  v4u g0;
  g0[0] = 1u;                                              // count=1, user mode
  g0[1] = lds_byte_off;                                    // lds_addr (bytes)
  g0[2] = (unsigned)(ga & 0xFFFFFFFFu);                    // global_addr[31:0]
  g0[3] = (unsigned)((ga >> 32) & 0x1FFFFFFu) | (2u << 30); // addr[56:32] | type=2

  const unsigned td0 = tile_d0;            // tensor_dim0 (elements)
  const unsigned td1 = rows_avail;         // tensor_dim1
  const unsigned long long s0 = row_stride;
  v8i g1;
  g1[0] = (int)(1u << 16);                                 // data_size=1 (2 bytes)
  g1[1] = (int)((td0 & 0xFFFFu) << 16);                    // tensor_dim0[15:0]
  g1[2] = (int)(((td0 >> 16) & 0xFFFFu) | ((td1 & 0xFFFFu) << 16));
  g1[3] = (int)(((td1 >> 16) & 0xFFFFu) | ((tile_d0 & 0xFFFFu) << 16)); // tile_dim0
  g1[4] = (int)(rows & 0xFFFFu);                           // tile_dim1 (tile_dim2=0)
  g1[5] = (int)(s0 & 0xFFFFFFFFull);                       // dim0_stride[31:0]
  g1[6] = (int)((s0 >> 32) & 0xFFFFull);                   // dim0_stride[47:32]
  g1[7] = 0;

  v4i z4 = {0, 0, 0, 0};
  v8i z8 = {0, 0, 0, 0, 0, 0, 0, 0};
  __builtin_amdgcn_tensor_load_to_lds(g0, g1, z4, z4, z8, 0);
}

// ---------------------------------------------------------------------------
// f32 -> f16 conversion (4 elements per thread)
// ---------------------------------------------------------------------------
__global__ __launch_bounds__(256) void f32_to_f16_kernel(const float* __restrict__ in,
                                                         _Float16* __restrict__ out, int n4) {
  int i = (blockIdx.x * blockDim.x + threadIdx.x) * 4;
  float4 v = *(const float4*)(in + i);
  _Float16* o = out + i;
  o[0] = (_Float16)v.x; o[1] = (_Float16)v.y; o[2] = (_Float16)v.z; o[3] = (_Float16)v.w;
}

// ---------------------------------------------------------------------------
// Generic GEMM: C[M,N] = A[M,K] @ B[N,K]^T, A/B f16, f32 accumulate.
// Block: 256 threads = 8 waves, 4(M) x 2(N); wave tile 32x32; WG tile 128x64.
// ---------------------------------------------------------------------------
template <typename OutT>
__global__ __launch_bounds__(256) void gemm_nt_wmma_h(const _Float16* __restrict__ A,
                                                      const _Float16* __restrict__ B,
                                                      OutT* __restrict__ C,
                                                      int M, int N, int K) {
  const int wave = threadIdx.x >> 5;
  const int wm = wave >> 1;
  const int wn = wave & 1;
  const int m0 = blockIdx.y * 128 + wm * 32;
  const int n0 = blockIdx.x * 64 + wn * 32;

  v8f acc00 = {}, acc01 = {}, acc10 = {}, acc11 = {};

  const _Float16* Am = A + (size_t)m0 * K;
  const _Float16* Bn = B + (size_t)n0 * K;

  for (int k = 0; k < K; k += 32) {
    __builtin_prefetch(Am + k + 256, 0, 1);     // global_prefetch_b8
    __builtin_prefetch(Bn + k + 256, 0, 1);
    v16h a0 = load_a_h(Am + k, K);
    v16h a1 = load_a_h(Am + (size_t)16 * K + k, K);
    v16h b0 = load_bT_h(B, K, n0, k);
    v16h b1 = load_bT_h(B, K, n0 + 16, k);
    acc00 = wmma_f16(a0, b0, acc00);
    acc01 = wmma_f16(a0, b1, acc01);
    acc10 = wmma_f16(a1, b0, acc10);
    acc11 = wmma_f16(a1, b1, acc11);
  }

  const int lane = threadIdx.x & 31;
  const int nn = lane & 15;
  const int mo = (lane < 16) ? 0 : 8;
#pragma unroll
  for (int r = 0; r < 8; ++r) {
    OutT* c0 = C + (size_t)(m0 + mo + r) * N + n0;
    OutT* c1 = C + (size_t)(m0 + 16 + mo + r) * N + n0;
    c0[nn]      = (OutT)acc00[r];
    c0[16 + nn] = (OutT)acc01[r];
    c1[nn]      = (OutT)acc10[r];
    c1[16 + nn] = (OutT)acc11[r];
  }
}

// ---------------------------------------------------------------------------
// RoPE in place on the f16 qkv workspace (math in f32).
// One thread per (b, s, h, pair); pairs are adjacent (2i, 2i+1).
// ---------------------------------------------------------------------------
__global__ __launch_bounds__(256) void rope_kernel(_Float16* __restrict__ qkv,
                                                   const float* __restrict__ fcos,
                                                   const float* __restrict__ fsin) {
  int idx = blockIdx.x * blockDim.x + threadIdx.x;   // B*S*H*64 total
  const int i = idx & 63;  idx >>= 6;
  const int h = idx & 15;  idx >>= 4;
  const int s = idx & 2047;
  const int b = idx >> 11;

  const float c  = fcos[s * 64 + i];
  const float sn = fsin[s * 64 + i];

  _Float16* q = qkv + (size_t)(b * SEQ + s) * LDQKV + h * HEAD_DIM + 2 * i;
  float t0 = (float)q[0], t1 = (float)q[1];
  q[0] = (_Float16)(t0 * c - t1 * sn);
  q[1] = (_Float16)(t0 * sn + t1 * c);

  _Float16* k = q + DIM;
  t0 = (float)k[0]; t1 = (float)k[1];
  k[0] = (_Float16)(t0 * c - t1 * sn);
  k[1] = (_Float16)(t0 * sn + t1 * c);
}

// ---------------------------------------------------------------------------
// Flash attention (causal). Grid: (S/64, B*H). Block: 128 threads = 4 waves.
// All waves share each 32-key step; K/V tiles are DMA'd to LDS by the TDM.
// Dynamic LDS layout: [0,8K) K tile 32x128 f16 | [8K,16K) V tile | [16K,20K) P
// ---------------------------------------------------------------------------
__global__ __launch_bounds__(128) void attn_kernel(const _Float16* __restrict__ qkv,
                                                   _Float16* __restrict__ attn_out) {
  extern __shared__ char smem[];
  _Float16* k_tile = (_Float16*)smem;                 // LDS byte offset 0
  _Float16* v_tile = (_Float16*)(smem + 8192);        // LDS byte offset 8192
  const int bh   = blockIdx.y;
  const int b    = bh >> 4;
  const int h    = bh & 15;
  const int wave = threadIdx.x >> 5;
  const int lane = threadIdx.x & 31;
  const int q0   = blockIdx.x * 64 + wave * 16;
  _Float16* pbuf = (_Float16*)(smem + 16384) + wave * 512;   // 16x32 probs tile

  const size_t ld = LDQKV;
  const _Float16* qbase = qkv + (size_t)(b * SEQ + q0) * ld + h * HEAD_DIM;
  const _Float16* kbase = qkv + (size_t)(b * SEQ) * ld + DIM     + h * HEAD_DIM;
  const _Float16* vbase = qkv + (size_t)(b * SEQ) * ld + 2 * DIM + h * HEAD_DIM;

  // Q: 16 rows x 128 head-dim as 4 A-frags (K = 32 each), kept in registers
  v16h qf[4];
#pragma unroll
  for (int t = 0; t < 4; ++t) qf[t] = load_a_h(qbase + t * 32, ld);

  v8f o[8];
#pragma unroll
  for (int t = 0; t < 8; ++t) o[t] = (v8f){};
  float mrow[8], lrow[8];
#pragma unroll
  for (int r = 0; r < 8; ++r) { mrow[r] = -1.0e30f; lrow[r] = 0.0f; }

  const int mo = (lane < 16) ? 0 : 8;
  const int nn = lane & 15;
  const float scale = 0.08838834764831845f;     // 1/sqrt(128)
  const int jend = blockIdx.x * 64 + 64;        // shared causal bound for the WG

  for (int j = 0; j < jend; j += 32) {
    __syncthreads();                            // previous tiles fully consumed
    if (wave == 0) {
      // DMA K and V 32x128 f16 tiles into LDS (rows beyond SEQ read as zero)
      tdm_load_tile_f16(0u,    kbase + (size_t)j * ld, HEAD_DIM, 32u,
                        (unsigned)(SEQ - j), (unsigned long long)ld);
      tdm_load_tile_f16(8192u, vbase + (size_t)j * ld, HEAD_DIM, 32u,
                        (unsigned)(SEQ - j), (unsigned long long)ld);
      __builtin_amdgcn_s_wait_tensorcnt(0);
    }
    __syncthreads();                            // tiles visible to all waves

    // ---- scores: S0 = Q x K[j..j+15]^T, S1 = Q x K[j+16..j+31]^T ----
    v8f s0 = {}, s1 = {};
#pragma unroll
    for (int t = 0; t < 4; ++t) {
      v16h bk0 = load_bT_h(k_tile, HEAD_DIM, 0,  t * 32);
      v16h bk1 = load_bT_h(k_tile, HEAD_DIM, 16, t * 32);
      s0 = wmma_f16(qf[t], bk0, s0);
      s1 = wmma_f16(qf[t], bk1, s1);
    }

    // ---- scale + causal mask + online softmax update ----
    float alpha[8];
#pragma unroll
    for (int r = 0; r < 8; ++r) {
      const int qrow = q0 + mo + r;
      float v0 = s0[r] * scale + (((j + nn) > qrow)      ? -1.0e9f : 0.0f);
      float v1 = s1[r] * scale + (((j + 16 + nn) > qrow) ? -1.0e9f : 0.0f);
      float mx = red_max16(fmaxf(v0, v1));
      float mn = fmaxf(mrow[r], mx);
      alpha[r] = __expf(mrow[r] - mn);
      mrow[r] = mn;
      float p0 = __expf(v0 - mn);
      float p1 = __expf(v1 - mn);
      lrow[r] = lrow[r] * alpha[r] + red_sum16(p0 + p1);
      pbuf[(mo + r) * 32 + nn]      = (_Float16)p0;   // C-layout -> row-major
      pbuf[(mo + r) * 32 + 16 + nn] = (_Float16)p1;
    }

#pragma unroll
    for (int t = 0; t < 8; ++t)
#pragma unroll
      for (int r = 0; r < 8; ++r) o[t][r] *= alpha[r];

    // ---- PV: O += P(16x32) x V[j..j+31, :] (P via LDS; in-order per wave) --
    v16h pa = load_a_h(pbuf, 32);
#pragma unroll
    for (int t = 0; t < 8; ++t) {
      v16h bv = load_b_kn_lds(v_tile, HEAD_DIM, t * 16);
      o[t] = wmma_f16(pa, bv, o[t]);
    }
  }

  // ---- normalize + store f16 (attn_out row-major [B*S, DIM]) ----
  _Float16* obase = attn_out + (size_t)(b * SEQ + q0) * DIM + h * HEAD_DIM;
#pragma unroll
  for (int r = 0; r < 8; ++r) {
    const float inv = 1.0f / lrow[r];
#pragma unroll
    for (int t = 0; t < 8; ++t)
      obase[(size_t)(mo + r) * DIM + t * 16 + nn] = (_Float16)(o[t][r] * inv);
  }
}

// ---------------------------------------------------------------------------
// Host-side launcher
// ---------------------------------------------------------------------------
extern "C" void kernel_launch(void* const* d_in, const int* in_sizes, int n_in,
                              void* d_out, int out_size, void* d_ws, size_t ws_size,
                              hipStream_t stream) {
  const float* x     = (const float*)d_in[0];
  const float* fcos  = (const float*)d_in[1];
  const float* fsin  = (const float*)d_in[2];
  // d_in[3] dense mask: causal -1e9 masking applied analytically.
  const float* w_in  = (const float*)d_in[4];
  const float* w_out = (const float*)d_in[5];
  float* out = (float*)d_out;

  const int M = BATCH * SEQ;                  // 4096
  const size_t n_x    = (size_t)M * DIM;      //  8.4M
  const size_t n_win  = (size_t)3 * DIM * DIM;// 12.6M
  const size_t n_wout = (size_t)DIM * DIM;    //  4.2M
  const size_t n_qkv  = (size_t)M * 3 * DIM;  // 25.2M

  _Float16* x_h    = (_Float16*)d_ws;
  _Float16* win_h  = x_h    + n_x;
  _Float16* wout_h = win_h  + n_win;
  _Float16* qkv_h  = wout_h + n_wout;
  _Float16* attn_h = qkv_h  + n_qkv;

  // 0) one-shot f32 -> f16 conversions of inputs
  f32_to_f16_kernel<<<(int)(n_x    / 1024), 256, 0, stream>>>(x,     x_h,    (int)n_x);
  f32_to_f16_kernel<<<(int)(n_win  / 1024), 256, 0, stream>>>(w_in,  win_h,  (int)n_win);
  f32_to_f16_kernel<<<(int)(n_wout / 1024), 256, 0, stream>>>(w_out, wout_h, (int)n_wout);

  // 1) qkv = x @ w_in^T   (f16 in, f32 accum, f16 out)
  {
    dim3 grid((3 * DIM) / 64, M / 128);
    gemm_nt_wmma_h<_Float16><<<grid, 256, 0, stream>>>(x_h, win_h, qkv_h, M, 3 * DIM, DIM);
  }

  // 2) RoPE on q and k sections in place
  {
    const int total = BATCH * SEQ * N_HEADS * (HEAD_DIM / 2);   // 4,194,304
    rope_kernel<<<total / 256, 256, 0, stream>>>(qkv_h, fcos, fsin);
  }

  // 3) causal flash attention per (b, h); TDM-fed LDS tiles
  {
    dim3 grid(SEQ / 64, BATCH * N_HEADS);
    attn_kernel<<<grid, 128, 20480, stream>>>(qkv_h, attn_h);
  }

  // 4) out = attn @ w_out^T  (f32 output)
  {
    dim3 grid(DIM / 64, M / 128);
    gemm_nt_wmma_h<float><<<grid, 256, 0, stream>>>(attn_h, wout_h, out, M, DIM, DIM);
  }
}
